// FractalDecoderV8_19645180412084
// MI455X (gfx1250) — matrix operand order repaired and checked by reading, not compiled, not run on hardware
//
#include <hip/hip_runtime.h>
#include <hip/hip_bf16.h>
#include <math.h>

// ---------------------------------------------------------------------------
// Types for CDNA5 WMMA (wave32): V_WMMA_F32_16X16X32_F16
// ---------------------------------------------------------------------------
typedef _Float16 v16h __attribute__((ext_vector_type(16)));
typedef _Float16 v8h  __attribute__((ext_vector_type(8)));
typedef float    v8f  __attribute__((ext_vector_type(8)));
typedef unsigned int u32x4 __attribute__((ext_vector_type(4)));
typedef int          i32x4 __attribute__((ext_vector_type(4)));
typedef int          i32x8 __attribute__((ext_vector_type(8)));

#define BROWS   4096
#define HID     256
#define INPAD   192      // 177 -> padded to multiple of 32
#define VOCABN  32000

// Fractal-kernel LDS map. Rows padded by 8 f16 (16B) so row strides are
// 132/260 DWORDs == 4 mod 64 banks -> conflict-free b128 fragment reads.
#define HS_STRIDE   264                    // 32 rows: 16,896 B
#define TS_STRIDE   520                    // 32 rows: 33,280 B
#define W1_STRIDE   264                    // 512 rows: 270,336 B
#define LDS_HS_OFF  0
#define LDS_TS_OFF  16896
#define LDS_W1_OFF  (16896 + 33280)        // 50,176
#define LDS_TOTAL   (50176 + 270336)       // 320,512 B <= 327,680 (320 KB)

__device__ __forceinline__ float gelu_exact(float v) {
    return 0.5f * v * (1.0f + erff(v * 0.70710678118654752f));
}

// Load a 16x32 (A) or 32x16 (B) f16 WMMA fragment from two contiguous
// 16-byte chunks (contiguous k order used here: element s of lane half h is k = 16h + s, same for A and B).
__device__ __forceinline__ v16h ld_frag(const _Float16* p0, const _Float16* p1) {
    v8h lo = *(const v8h*)p0;
    v8h hi = *(const v8h*)p1;
    v16h r;
#pragma unroll
    for (int i = 0; i < 8; ++i) { r[i] = lo[i]; r[i + 8] = hi[i]; }
    return r;
}

// ---------------------------------------------------------------------------
// TDM: stage a 2-D f16 tile (rows x rowlen elems, row stride 'stride' elems)
// from global into LDS at byte offset lds_off, optionally inserting LDS row
// padding (pad_interval encoded per ISA: 1<<pi 8-byte units between pads;
// pad_amount_dw: (value+1) DWORDs of pad). One DMA, tracked by TENSORcnt.
// ---------------------------------------------------------------------------
__device__ __forceinline__ void tdm_load_2d(const _Float16* gsrc, unsigned lds_off,
                                            unsigned rowlen, unsigned rows,
                                            unsigned stride,
                                            int pad_en, unsigned pi, unsigned pa) {
    unsigned long long ga = (unsigned long long)(uintptr_t)gsrc;
    u32x4 g0;
    g0.x = 1u;                                  // count=1, user descriptor
    g0.y = lds_off;                             // lds_addr (bytes)
    g0.z = (unsigned)(ga & 0xFFFFFFFFu);        // global_addr[31:0]
    g0.w = (unsigned)((ga >> 32) & 0x01FFFFFFu) // global_addr[56:32]
           | (2u << 30);                        // type=2 ("image")
    i32x8 g1;
    g1[0] = (int)((1u << 16)                    // data_size=1 -> 2 bytes
           | (pad_en ? (1u << 20) : 0u)         // pad_enable
           | ((pi & 7u) << 22)                  // pad_interval
           | ((pa & 127u) << 25));              // pad_amount (DWORDs-1)
    g1[1] = (int)((rowlen & 0xFFFFu) << 16);    // tensor_dim0[15:0]
    g1[2] = (int)((rowlen >> 16) | ((rows & 0xFFFFu) << 16));  // dim0 hi | dim1 lo
    g1[3] = (int)((rows >> 16) | (rowlen << 16));              // dim1 hi | tile_dim0
    g1[4] = (int)rows;                          // tile_dim1 (tile_dim2=0)
    g1[5] = (int)stride;                        // tensor_dim0_stride[31:0]
    g1[6] = 0;                                  // stride hi | dim1_stride lo
    g1[7] = 0;
    i32x4 z4 = {0, 0, 0, 0};
#if defined(__clang_major__) && (__clang_major__ >= 23)
    i32x8 z8 = {0, 0, 0, 0, 0, 0, 0, 0};
    __builtin_amdgcn_tensor_load_to_lds(g0, g1, z4, z4, z8, 0);
#else
    __builtin_amdgcn_tensor_load_to_lds(g0, g1, z4, z4, 0);
#endif
}

// ---------------------------------------------------------------------------
// Weight prep: fp32 [K,N] row-major  ->  f16 [N,Kp] ("B transposed"), zero-pad K.
// ---------------------------------------------------------------------------
__global__ void __launch_bounds__(256)
cast_transpose_kernel(const float* __restrict__ src, _Float16* __restrict__ dst,
                      int K, int N, int Kp) {
    int i = blockIdx.x * 256 + threadIdx.x;
    int total = N * Kp;
    if (i >= total) return;
    int n = i / Kp, k = i - n * Kp;
    dst[i] = (k < K) ? (_Float16)src[(size_t)k * N + n] : (_Float16)0.0f;
}

// ---------------------------------------------------------------------------
// Featurize: fourier (1+96) + hashgrid (48) + latent (32) -> f16 [B,192]
// ---------------------------------------------------------------------------
__global__ void __launch_bounds__(192)
featurize_kernel(const float* __restrict__ x, const float* __restrict__ z,
                 const float* __restrict__ tables,  // [12,16384,4]
                 _Float16* __restrict__ feats) {    // [B,192]
    const int row = blockIdx.x;
    const int tid = threadIdx.x;
    const float xv = x[row];
    float v = 0.0f;
    if (tid == 0) {
        v = xv;
    } else if (tid <= 96) {
        int i = (tid <= 48) ? (tid - 1) : (tid - 49);
        float freq = expf((float)i * (4.0f / 47.0f) * 2.302585092994046f);
        float arg = 6.28318530717958647f * xv * freq;
        v = (tid <= 48) ? sinf(arg) : cosf(arg);
    } else if (tid < 145) {
        int idx = tid - 97;
        int l = idx >> 2, d = idx & 3;
        float res = 16.0f * expf((float)l * 0.405465108108164f); // 1.5^l
        float pos = xv * res;
        float fl = floorf(pos);
        unsigned il = (unsigned)fl;
        unsigned hl = (il * 2654435761u) & 16383u;
        unsigned hr = ((il + 1u) * 2654435761u) & 16383u;
        float t = fminf(fmaxf(pos - fl, 0.0f), 1.0f);
        float el = tables[((size_t)l * 16384u + hl) * 4u + d];
        float er = tables[((size_t)l * 16384u + hr) * 4u + d];
        v = (1.0f - t) * el + t * er;
    } else if (tid < 177) {
        v = z[(size_t)row * 32 + (tid - 145)];
    } // 177..191: zero pad
    feats[(size_t)row * INPAD + tid] = (_Float16)v;
}

// ---------------------------------------------------------------------------
// Generic WMMA GEMM: C = act(A[M,Kp] @ Bt[N,Kp]^T + bias)
//   256 threads = 8 waves; block tile 64(M) x 128(N); wave tile 16 x 64.
//   ACT==1: exact GELU -> f16 out;  ACT==0: none -> fp32 nontemporal out.
// ---------------------------------------------------------------------------
template <int ACT>
__global__ void __launch_bounds__(256)
gemm_bias_act(const _Float16* __restrict__ A,   // [M,Kp] row-major
              const _Float16* __restrict__ Bt,  // [N,Kp]
              const float* __restrict__ bias,   // [N]
              void* __restrict__ Cout, int Kp, int N) {
    const int lane = threadIdx.x & 31;
    const int wave = threadIdx.x >> 5;
    const int lr = lane & 15, lh = lane >> 4;
    const int m0 = blockIdx.x * 64 + (wave >> 1) * 16;
    const int nb = blockIdx.y * 128 + (wave & 1) * 64;

    const _Float16* Arow = A + (size_t)(m0 + lr) * Kp;
    v8f acc[4] = {};
    for (int k = 0; k < Kp; k += 32) {
        __builtin_prefetch(Arow + k + 64, 0, 1);  // global_prefetch_b8
        v16h a = ld_frag(Arow + k + lh * 16, Arow + k + lh * 16 + 8);
        v16h bfr[4];
#pragma unroll
        for (int j = 0; j < 4; ++j) {
            const _Float16* bp = Bt + (size_t)(nb + j * 16 + lr) * Kp + k + lh * 16;
            bfr[j] = ld_frag(bp, bp + 8);
        }
#pragma unroll
        for (int j = 0; j < 4; ++j)
            acc[j] = __builtin_amdgcn_wmma_f32_16x16x32_f16(
                false, a, false, bfr[j], (short)0, acc[j], false, false);
    }
#pragma unroll
    for (int j = 0; j < 4; ++j) {
        int n = nb + j * 16 + lr;
        float bv = bias[n];
#pragma unroll
        for (int r = 0; r < 8; ++r) {
            int m = m0 + r + 8 * lh;  // D layout: VGPR r -> M = r + 8*(lane>=16)
            float v = acc[j][r] + bv;
            if (ACT == 1)
                ((_Float16*)Cout)[(size_t)m * N + n] = (_Float16)gelu_exact(v);
            else
                __builtin_nontemporal_store(v, &((float*)Cout)[(size_t)m * N + n]);
        }
    }
}

// ---------------------------------------------------------------------------
// Fractal loop: 64 x ( h = 0.5h + 0.5*tanh(gelu(h@W1+b1)@W2+b2) )
// Block owns 32 rows. LDS (bank-conflict-free padded rows):
//   h (16.5KB) + t (32.5KB) + FULL W1 (264KB, TDM-staged once with LDS row
//   padding, resident across all iterations) = 313KB of 320KB WGP LDS.
// W2 (256KB f16) streams from L2 per fragment.
// ---------------------------------------------------------------------------
__global__ void __launch_bounds__(256)
fractal_kernel(_Float16* __restrict__ Hg,        // [B,256] in/out
               const _Float16* __restrict__ W1t, // [512,256]
               const _Float16* __restrict__ W2t, // [256,512]
               const float* __restrict__ b1,     // [512]
               const float* __restrict__ b2) {   // [256]
    extern __shared__ _Float16 smem[];
    _Float16* hs = smem;                        // [32][HS_STRIDE]
    _Float16* ts = smem + LDS_TS_OFF / 2;       // [32][TS_STRIDE]
    _Float16* w1 = smem + LDS_W1_OFF / 2;       // [512][W1_STRIDE]

    const int tid  = threadIdx.x;
    const int lane = tid & 31, wave = tid >> 5;
    const int lr = lane & 15, lh = lane >> 4;
    const int row0 = blockIdx.x * 32;
    const int m0 = (wave >> 2) * 16;            // 2 local M tiles (32 rows)

    // One-shot TDM stage of W1 [512 x 256 f16] into padded LDS image:
    // rows are 512B = 64 x 8B units -> pad_interval=6; pad 4 DWORDs -> pa=3.
    if (wave == 0) {
        tdm_load_2d(W1t, LDS_W1_OFF, /*rowlen=*/HID, /*rows=*/512, /*stride=*/HID,
                    /*pad_en=*/1, /*pi=*/6, /*pa=*/3);
    }
    // Meanwhile: stage h tile cooperatively into padded rows (b128 chunks).
    for (int i = tid; i < 32 * 32; i += 256) {   // 32 rows x 32 chunks of 8 f16
        int r = i >> 5, c = (i & 31) * 8;
        *(v8h*)(hs + r * HS_STRIDE + c) = *(const v8h*)(Hg + (size_t)(row0 + r) * 256 + c);
    }
    if (wave == 0) __builtin_amdgcn_s_wait_tensorcnt(0);
    __syncthreads();   // publishes both W1 (TDM) and h to the workgroup

    for (int it = 0; it < 64; ++it) {
        // ---- phase 1: ts = gelu(hs @ W1 + b1), [32,512], K=256, B from LDS ----
        {
            const int nwb = (wave & 3) * 128;   // each wave: 128 cols = 8 tiles
            v8f acc[8] = {};
            for (int k = 0; k < 256; k += 32) {
                const _Float16* ap = hs + (m0 + lr) * HS_STRIDE + k;
                v16h a = ld_frag(ap + lh * 16, ap + lh * 16 + 8);
                v16h bfr[8];
#pragma unroll
                for (int j = 0; j < 8; ++j) {
                    const _Float16* bp = w1 + (nwb + j * 16 + lr) * W1_STRIDE + k + lh * 16;
                    bfr[j] = ld_frag(bp, bp + 8);
                }
#pragma unroll
                for (int j = 0; j < 8; ++j)
                    acc[j] = __builtin_amdgcn_wmma_f32_16x16x32_f16(
                        false, a, false, bfr[j], (short)0, acc[j], false, false);
            }
#pragma unroll
            for (int j = 0; j < 8; ++j) {
                int n = nwb + j * 16 + lr;
                float bv = b1[n];
#pragma unroll
                for (int r = 0; r < 8; ++r) {
                    int m = m0 + r + 8 * lh;
                    ts[m * TS_STRIDE + n] = (_Float16)gelu_exact(acc[j][r] + bv);
                }
            }
        }
        __syncthreads();

        // ---- phase 2: u = tanh(ts @ W2 + b2); hs = 0.5hs + 0.5u, K=512 ----
        {
            const int nwb = (wave & 3) * 64;    // each wave: 64 cols = 4 tiles
            v8f acc[4] = {};
            for (int k = 0; k < 512; k += 32) {
                const _Float16* ap = ts + (m0 + lr) * TS_STRIDE + k;
                v16h a = ld_frag(ap + lh * 16, ap + lh * 16 + 8);
                v16h bfr[4];
#pragma unroll
                for (int j = 0; j < 4; ++j) {
                    const _Float16* bp = W2t + (size_t)(nwb + j * 16 + lr) * 512 + k + lh * 16;
                    bfr[j] = ld_frag(bp, bp + 8);
                }
#pragma unroll
                for (int j = 0; j < 4; ++j)
                    acc[j] = __builtin_amdgcn_wmma_f32_16x16x32_f16(
                        false, a, false, bfr[j], (short)0, acc[j], false, false);
            }
#pragma unroll
            for (int j = 0; j < 4; ++j) {
                int n = nwb + j * 16 + lr;
                float bv = b2[n];
#pragma unroll
                for (int r = 0; r < 8; ++r) {
                    int m = m0 + r + 8 * lh;
                    float u = tanhf(acc[j][r] + bv);
                    float hv = (float)hs[m * HS_STRIDE + n]; // (m,n) owned by this wave
                    hs[m * HS_STRIDE + n] = (_Float16)(0.5f * hv + 0.5f * u);
                }
            }
        }
        __syncthreads();
    }

    // write h back (de-pad rows)
    for (int i = tid; i < 32 * 32; i += 256) {
        int r = i >> 5, c = (i & 31) * 8;
        *(v8h*)(Hg + (size_t)(row0 + r) * 256 + c) = *(const v8h*)(hs + r * HS_STRIDE + c);
    }
}

// ---------------------------------------------------------------------------
// Host-side orchestration
// ---------------------------------------------------------------------------
extern "C" void kernel_launch(void* const* d_in, const int* in_sizes, int n_in,
                              void* d_out, int out_size, void* d_ws, size_t ws_size,
                              hipStream_t stream) {
    (void)in_sizes; (void)n_in; (void)out_size; (void)ws_size;
    const float* x      = (const float*)d_in[0];
    const float* z      = (const float*)d_in[1];
    const float* tables = (const float*)d_in[2];
    const float* W_in1  = (const float*)d_in[3];
    const float* b_in1  = (const float*)d_in[4];
    const float* W_in2  = (const float*)d_in[5];
    const float* b_in2  = (const float*)d_in[6];
    const float* W_f1   = (const float*)d_in[7];
    const float* b_f1   = (const float*)d_in[8];
    const float* W_f2   = (const float*)d_in[9];
    const float* b_f2   = (const float*)d_in[10];
    const float* W_out  = (const float*)d_in[11];
    const float* b_out  = (const float*)d_in[12];
    float* out = (float*)d_out;

    char* ws = (char*)d_ws;
    size_t off = 0;
    auto alloc = [&](size_t bytes) {
        char* p = ws + off;
        off = (off + bytes + 255) & ~(size_t)255;
        return p;
    };
    _Float16* feats = (_Float16*)alloc((size_t)BROWS * INPAD * 2);
    _Float16* Wi1t  = (_Float16*)alloc((size_t)HID * INPAD * 2);      // [256,192]
    _Float16* Wi2t  = (_Float16*)alloc((size_t)HID * HID * 2);        // [256,256]
    _Float16* Wf1t  = (_Float16*)alloc((size_t)512 * HID * 2);        // [512,256]
    _Float16* Wf2t  = (_Float16*)alloc((size_t)HID * 512 * 2);        // [256,512]
    _Float16* Wot   = (_Float16*)alloc((size_t)VOCABN * HID * 2);     // [32000,256]
    _Float16* h1    = (_Float16*)alloc((size_t)BROWS * HID * 2);
    _Float16* h     = (_Float16*)alloc((size_t)BROWS * HID * 2);

    // weight prep (f16, transposed, K padded to 32)
    cast_transpose_kernel<<<(HID * INPAD + 255) / 256, 256, 0, stream>>>(W_in1, Wi1t, 177, HID, INPAD);
    cast_transpose_kernel<<<(HID * HID + 255) / 256, 256, 0, stream>>>(W_in2, Wi2t, HID, HID, HID);
    cast_transpose_kernel<<<(512 * HID + 255) / 256, 256, 0, stream>>>(W_f1, Wf1t, HID, 512, HID);
    cast_transpose_kernel<<<(HID * 512 + 255) / 256, 256, 0, stream>>>(W_f2, Wf2t, 512, HID, 512);
    cast_transpose_kernel<<<((size_t)VOCABN * HID + 255) / 256, 256, 0, stream>>>(W_out, Wot, HID, VOCABN, HID);

    // featurize
    featurize_kernel<<<BROWS, 192, 0, stream>>>(x, z, tables, feats);

    // input MLP: h = gelu(gelu(feats@W_in1+b) @ W_in2 + b)
    gemm_bias_act<1><<<dim3(BROWS / 64, HID / 128), 256, 0, stream>>>(feats, Wi1t, b_in1, (void*)h1, INPAD, HID);
    gemm_bias_act<1><<<dim3(BROWS / 64, HID / 128), 256, 0, stream>>>(h1, Wi2t, b_in2, (void*)h, HID, HID);

    // 64 weight-tied fractal steps; W1 LDS-resident via padded TDM stage
    fractal_kernel<<<BROWS / 32, 256, LDS_TOTAL, stream>>>(h, Wf1t, Wf2t, b_f1, b_f2);

    // output projection: [4096,32000] fp32 (store-bound, nontemporal)
    gemm_bias_act<0><<<dim3(BROWS / 64, VOCABN / 128), 256, 0, stream>>>(h, Wot, b_out, (void*)out, HID, VOCABN);
}
